// Quantizer_19267223290735
// MI455X (gfx1250) — matrix operand order repaired and checked by reading, not compiled, not run on hardware
//
#include <hip/hip_runtime.h>

// ---------------------------------------------------------------------------
// VQ-VAE quantizer for gfx1250 (MI455X).
//   B=32, D=64, H=W=64  -> N = 131072 positions, K = 512 codes.
//   Core: scores = z @ cb^T via v_wmma_f32_16x16x32_f16, C seeded with
//         -0.5*||e||^2 so argmax(score) == argmin(distance).
// ---------------------------------------------------------------------------

typedef _Float16 half_t;
typedef __attribute__((ext_vector_type(16))) _Float16 v16h;
typedef __attribute__((ext_vector_type(8)))  _Float16 v8h;
typedef __attribute__((ext_vector_type(8)))  float    v8f;

union Frag16 { v16h v; v8h h[2]; };

#define N_POS     131072   // 32*64*64
#define K_CODES   512
#define D_DIM     64
#define HW        4096     // 64*64
#define ROWS_WG   128      // positions per workgroup
#define LDS_STRIDE 72      // halves per row: 144 B, 16B-aligned chunks

// ---------------------------------------------------------------------------
// Kernel 1: codebook fp32 -> f16 copy + per-row squared norms.
// grid = 512 blocks, 64 threads (one codebook row per block).
// ---------------------------------------------------------------------------
__global__ __launch_bounds__(64)
void vq_prep_kernel(const float* __restrict__ cb,
                    half_t* __restrict__ cb16,
                    float* __restrict__ norms) {
    const int r = blockIdx.x;
    const int t = threadIdx.x;
    const float v = cb[r * D_DIM + t];
    cb16[r * D_DIM + t] = (half_t)v;
    __shared__ float s[D_DIM];
    s[t] = v * v;
    __syncthreads();
    if (t == 0) {
        float a = 0.f;
        for (int i = 0; i < D_DIM; ++i) a += s[i];
        norms[r] = a;
    }
}

// ---------------------------------------------------------------------------
// Kernel 2: WMMA distance-GEMM + argmin.
// grid = 1024 workgroups x 128 threads (4 waves); each wave owns 32 rows.
// ---------------------------------------------------------------------------
__global__ __launch_bounds__(128)
void vq_argmin_kernel(const float* __restrict__ z,        // [32][64][4096] NCHW
                      const half_t* __restrict__ cb16,    // [512][64]
                      const float* __restrict__ norms,    // [512]
                      int* __restrict__ best_idx) {       // [N_POS]
    __shared__ __align__(16) half_t zt[ROWS_WG * LDS_STRIDE]; // [pos][k] f16

    const int t   = threadIdx.x;                // 0..127
    const int p0  = blockIdx.x * ROWS_WG;       // 128 | 4096 -> single image
    const int b   = p0 >> 12;
    const int hw0 = p0 & (HW - 1);

    // ---- stage z tile into LDS, transposed to [pos][k], f16 ----
    // coalesced: 128 consecutive hw per k-slice.
    const float* zb = z + (size_t)b * D_DIM * HW + hw0 + t;
    #pragma unroll 8
    for (int k = 0; k < D_DIM; ++k)
        zt[t * LDS_STRIDE + k] = (half_t)zb[(size_t)k * HW];
    __syncthreads();

    const int wave     = t >> 5;       // wave32
    const int lane     = t & 31;
    const int half_sel = lane >> 4;    // 0: lanes 0-15, 1: lanes 16-31
    const int lr       = lane & 15;
    const int kb       = half_sel * 16; // A-layout K base per half

    // ---- A fragments for 2 row-blocks of 16 (this code's contiguous k order, k = 16h + s) ----
    Frag16 a_lo[2], a_hi[2];
    #pragma unroll
    for (int rb = 0; rb < 2; ++rb) {
        const half_t* row = &zt[(wave * 32 + rb * 16 + lr) * LDS_STRIDE];
        a_lo[rb].h[0] = *(const v8h*)(row + kb);        // K kb..kb+7
        a_lo[rb].h[1] = *(const v8h*)(row + kb + 8);    // K kb+8..kb+15
        a_hi[rb].h[0] = *(const v8h*)(row + kb + 32);
        a_hi[rb].h[1] = *(const v8h*)(row + kb + 40);
    }

    float best[2][8];
    int   bidx[2][8];
    #pragma unroll
    for (int rb = 0; rb < 2; ++rb)
        #pragma unroll
        for (int v = 0; v < 8; ++v) { best[rb][v] = -3.4e38f; bidx[rb][v] = 0; }

    const int kbase2 = half_sel * 16;  // contiguous k as in A: lanes>=16 hold K 16..31

    // ---- sweep the 512 codes in 32 N-tiles of 16 ----
    for (int tn = 0; tn < 32; ++tn) {
        const int c = tn * 16 + lr;                 // this lane's code column
        const half_t* cbp = cb16 + (size_t)c * D_DIM + kbase2;
        Frag16 blo, bhi;
        blo.h[0] = *(const v8h*)(cbp);              // 16 contiguous halves
        blo.h[1] = *(const v8h*)(cbp + 8);
        bhi.h[0] = *(const v8h*)(cbp + 32);         // K 32..63 half
        bhi.h[1] = *(const v8h*)(cbp + 40);
        const float initc = -0.5f * norms[c];

        #pragma unroll
        for (int rb = 0; rb < 2; ++rb) {
            v8f acc;
            #pragma unroll
            for (int v = 0; v < 8; ++v) acc[v] = initc;   // C seeded -0.5||e||^2
            acc = __builtin_amdgcn_wmma_f32_16x16x32_f16(
                      false, a_lo[rb].v, false, blo.v, (short)0, acc, false, false);
            acc = __builtin_amdgcn_wmma_f32_16x16x32_f16(
                      false, a_hi[rb].v, false, bhi.v, (short)0, acc, false, false);
            #pragma unroll
            for (int v = 0; v < 8; ++v) {
                if (acc[v] > best[rb][v]) { best[rb][v] = acc[v]; bidx[rb][v] = c; }
            }
        }
    }

    // ---- argmax butterfly inside each 16-lane half; min-index tie-break ----
    #pragma unroll
    for (int rb = 0; rb < 2; ++rb) {
        #pragma unroll
        for (int v = 0; v < 8; ++v) {
            float b0 = best[rb][v];
            int   i0 = bidx[rb][v];
            #pragma unroll
            for (int off = 1; off < 16; off <<= 1) {
                float ob = __shfl_xor(b0, off, 32);
                int   oi = __shfl_xor(i0, off, 32);
                if (ob > b0 || (ob == b0 && oi < i0)) { b0 = ob; i0 = oi; }
            }
            if (lr == 0) {
                // C layout: lane half holds rows v (+8 for lanes 16-31)
                const int m = wave * 32 + rb * 16 + half_sel * 8 + v;
                best_idx[p0 + m] = i0;
            }
        }
    }
}

// ---------------------------------------------------------------------------
// Kernel 3: gather exact fp32 codebook rows, write NCHW quantized + losses.
// grid = 4096 blocks, block = (32 hw, 8 dim-groups).
// ---------------------------------------------------------------------------
__global__ __launch_bounds__(256)
void vq_epilogue_kernel(const float* __restrict__ z,
                        const float* __restrict__ cb,
                        const int* __restrict__ idx,
                        float* __restrict__ qout,
                        float* __restrict__ loss,
                        float* __restrict__ commit,
                        float* __restrict__ embed) {
    const int tx = threadIdx.x;                 // hw within chunk
    const int ty = threadIdx.y;                 // dim group
    const int p  = blockIdx.x * 32 + tx;
    const int b  = p >> 12;
    const int hw = p & (HW - 1);
    const int k  = idx[p];
    const float* cbr = cb + (size_t)k * D_DIM;

    const size_t basez = (size_t)b * D_DIM * HW + hw;
    float acc = 0.f;
    #pragma unroll
    for (int i = 0; i < 8; ++i) {
        const int d = ty * 8 + i;
        const size_t a = basez + (size_t)d * HW;
        const float zv = z[a];
        const float q  = cbr[d];
        qout[a] = q;                            // straight-through == quantized
        const float df = q - zv;
        acc += df * df;
    }

    __shared__ float part[8][33];
    part[ty][tx] = acc;
    __syncthreads();
    if (ty == 0) {
        float s = 0.f;
        #pragma unroll
        for (int j = 0; j < 8; ++j) s += part[j][tx];
        const float mse = s * (1.0f / 64.0f);
        loss[p]   = 1.25f * mse;   // 0.25*mse + mse
        commit[p] = mse;
        embed[p]  = mse;
    }
}

// ---------------------------------------------------------------------------
extern "C" void kernel_launch(void* const* d_in, const int* in_sizes, int n_in,
                              void* d_out, int out_size, void* d_ws, size_t ws_size,
                              hipStream_t stream) {
    const float* z  = (const float*)d_in[0];   // [32,64,64,64] fp32 NCHW
    const float* cb = (const float*)d_in[1];   // [512,64] fp32

    char* ws = (char*)d_ws;
    half_t* cb16 = (half_t*)ws;                         // 64 KB
    float*  norms = (float*)(ws + 65536);               // 2 KB
    int*    idx   = (int*)(ws + 68608);                 // 512 KB

    float* out    = (float*)d_out;
    float* qout   = out;                                // 8388608
    float* loss   = out + 8388608;                      // 131072
    float* commit = loss + N_POS;                       // 131072
    float* embed  = commit + N_POS;                     // 131072

    vq_prep_kernel<<<K_CODES, D_DIM, 0, stream>>>(cb, cb16, norms);
    vq_argmin_kernel<<<N_POS / ROWS_WG, ROWS_WG, 0, stream>>>(z, cb16, norms, idx);
    vq_epilogue_kernel<<<N_POS / 32, dim3(32, 8), 0, stream>>>(
        z, cb, idx, qout, loss, commit, embed);
}
